// CharStemmer_44229573214684
// MI455X (gfx1250) — compile-verified, matrix operand order checked by reading, not run on hardware
//
#include <hip/hip_runtime.h>
#include <cstdint>
#include <cstddef>

// ---------------------------------------------------------------------------
// Seq2seq LSTM (bi-encoder + decoder + vocab projection) for gfx1250 (MI455X)
// bf16 WMMA (v_wmma_f32_16x16x32_bf16) GEMMs, f32 gate math / cell state.
// Big GEMMs: LDS-staged (async global->LDS when available), double-buffered.
// ---------------------------------------------------------------------------

typedef __bf16 bf16;
typedef __attribute__((ext_vector_type(16))) __bf16 v16bf;
typedef __attribute__((ext_vector_type(8)))  __bf16 v8bf;
typedef __attribute__((ext_vector_type(8)))  float  v8f;

#define SEQ 128
#define BAT 128
#define VOC 61
#define VOCP 64            // vocab padded to WMMA tile multiple
#define EMB 512
#define HID 1024
#define G4  4096           // 4*HID gates
#define SB  (SEQ * BAT)

// LDS tile geometry for the big GEMM
#define BM 64
#define BN 256
#define BK 32
#define LDSS 40            // padded row stride (elements): 80B -> 16B aligned,
                           // 20-bank stride => conflict-free b128 reads

#if __has_builtin(__builtin_amdgcn_global_load_async_to_lds_b128) && \
    __has_builtin(__builtin_amdgcn_s_wait_asynccnt)
#define USE_ASYNC_LDS 1
#else
#define USE_ASYNC_LDS 0
#endif

// ---------------------------------------------------------------------------
// WMMA helpers
// ---------------------------------------------------------------------------

// 16x32 bf16 A/B fragment from a row-major matrix: lane (l&15) owns row r,
// ISA 7.12.2 16-bit A layout: lanes 0-15 hold K={0..7,16..23}, lanes 16-31
// hold K={8..15,24..31}.  rowp points at element (row, 0); 16B aligned.
__device__ __forceinline__ v16bf frag16(const bf16* __restrict__ rowp, int k0, int lane) {
  const int h8 = (lane & 16) >> 1;  // lanes 0-15 -> 0 ; lanes 16-31 -> 8
  const v8bf lo = *reinterpret_cast<const v8bf*>(rowp + k0 + h8);
  const v8bf hi = *reinterpret_cast<const v8bf*>(rowp + k0 + 16 + h8);
  v16bf f;
#pragma unroll
  for (int i = 0; i < 8; ++i) { f[i] = lo[i]; f[i + 8] = hi[i]; }
  return f;
}

__device__ __forceinline__ v8f wmma_bf(v16bf a, v16bf b, v8f c) {
  return __builtin_amdgcn_wmma_f32_16x16x32_bf16(false, a, false, b, (short)0, c,
                                                 false, false);
}

__device__ __forceinline__ float sigm(float x) { return 1.0f / (1.0f + __expf(-x)); }

// ---------------------------------------------------------------------------
// Tile staging: 16B per call, global -> LDS.
// Async path: GLOBAL_LOAD_ASYNC_TO_LDS_B128 (ASYNCcnt tracked, no VGPR data).
// Builtin signature (from hipcc diagnostic): param0 = v4i* in the global
// address space, so pass explicitly AS-qualified v4i pointers.
// Fallback:   reg copy (global_load_b128 + ds_store_b128).
// ---------------------------------------------------------------------------
typedef __attribute__((vector_size(16))) int v4i;
#if USE_ASYNC_LDS
typedef __attribute__((address_space(1))) v4i* gptr_v4i;
typedef __attribute__((address_space(3))) v4i* lptr_v4i;
#endif

__device__ __forceinline__ void stage16(const bf16* __restrict__ g, bf16* __restrict__ l) {
#if USE_ASYNC_LDS
  // flat LDS address low 32 bits == LDS byte offset (aperture truncation rule)
  __builtin_amdgcn_global_load_async_to_lds_b128(
      (gptr_v4i)(uintptr_t)g, (lptr_v4i)(uint32_t)(uintptr_t)l, 0, 0);
#else
  *reinterpret_cast<v8bf*>(l) = *reinterpret_cast<const v8bf*>(g);
#endif
}

// Stage one BK-chunk: A tile [BM x BK] and B tile [BN x BK] (rows padded to
// LDSS).  256 threads, 5 x 16B per thread => 5 async ops per wave per chunk.
__device__ __forceinline__ void stage_chunk(const bf16* __restrict__ A, int lda,
                                            const bf16* __restrict__ W, int ldb,
                                            int m0, int n0, int kc, int tid,
                                            bf16* __restrict__ As, bf16* __restrict__ Bs) {
  const int r  = tid >> 2;         // 0..63
  const int cg = (tid & 3) * 8;    // 0,8,16,24
  stage16(A + (size_t)(m0 + r) * lda + kc + cg, As + r * LDSS + cg);
#pragma unroll
  for (int it = 0; it < 4; ++it) {
    const int rr = r + it * 64;    // 0..255
    stage16(W + (size_t)(n0 + rr) * ldb + kc + cg, Bs + rr * LDSS + cg);
  }
}

// ---------------------------------------------------------------------------
// Big GEMM + bias:  C[M,N] = A[M,K](bf16) @ W[N,K]^T(bf16) + bias[N]
// Block 256 thr = 8 waves (2x4), block tile 64x256, wave tile 32x64.
// grid = (M/64, N/256).  N, M multiples of tile; K multiple of 32.
// ---------------------------------------------------------------------------
__global__ __launch_bounds__(256) void k_gemm_bias_big(
    const bf16* __restrict__ A, int lda,
    const bf16* __restrict__ W, int ldb,
    const float* __restrict__ bias,
    float* __restrict__ C, int ldc, int K) {
  __shared__ bf16 As[2][BM * LDSS];
  __shared__ bf16 Bs[2][BN * LDSS];

  const int tid  = threadIdx.x;
  const int lane = tid & 31;
  const int wid  = tid >> 5;
  const int m0   = blockIdx.x * BM;
  const int n0   = blockIdx.y * BN;
  const int moff = (wid >> 2) * 32;   // 0,32
  const int noff = (wid & 3) * 64;    // 0,64,128,192

  v8f acc[2][4] = {};

  auto compute = [&](const bf16* __restrict__ Ab, const bf16* __restrict__ Bb) {
    const bf16* ar0 = Ab + (moff + (lane & 15)) * LDSS;
    const v16bf a0 = frag16(ar0, 0, lane);
    const v16bf a1 = frag16(ar0 + 16 * LDSS, 0, lane);
#pragma unroll
    for (int j = 0; j < 4; ++j) {
      const bf16* br = Bb + (noff + j * 16 + (lane & 15)) * LDSS;
      const v16bf b = frag16(br, 0, lane);
      acc[0][j] = wmma_bf(a0, b, acc[0][j]);
      acc[1][j] = wmma_bf(a1, b, acc[1][j]);
    }
  };

  const int nk = K / BK;
  stage_chunk(A, lda, W, ldb, m0, n0, 0, tid, As[0], Bs[0]);
  for (int k = 0; k < nk - 1; ++k) {
    const int cur = k & 1;
    stage_chunk(A, lda, W, ldb, m0, n0, (k + 1) * BK, tid, As[cur ^ 1], Bs[cur ^ 1]);
#if USE_ASYNC_LDS
    __builtin_amdgcn_s_wait_asynccnt(5);  // chunk k's 5 ops done; k+1 in flight
#endif
    __syncthreads();
    compute(As[cur], Bs[cur]);
    __syncthreads();
  }
#if USE_ASYNC_LDS
  __builtin_amdgcn_s_wait_asynccnt(0);
#endif
  __syncthreads();
  compute(As[(nk - 1) & 1], Bs[(nk - 1) & 1]);

  const int lm = (lane >> 4) << 3;
#pragma unroll
  for (int i = 0; i < 2; ++i) {
#pragma unroll
    for (int j = 0; j < 4; ++j) {
      const int n  = n0 + noff + j * 16 + (lane & 15);
      const float bb = bias[n];
      const int mb = m0 + moff + i * 16 + lm;
#pragma unroll
      for (int v = 0; v < 8; ++v)
        C[(size_t)(mb + v) * ldc + n] = acc[i][j][v] + bb;
    }
  }
}

// ---------------------------------------------------------------------------
// Small GEMM + bias (vocab projection, N=61):  direct-load, 16x16 per wave.
// grid.x = M/32, grid.y = ceil(N/64).  W zero-padded to ceil(N/16)*16 rows.
// ---------------------------------------------------------------------------
__global__ __launch_bounds__(256) void k_gemm_bias_small(
    const bf16* __restrict__ A, int lda,
    const bf16* __restrict__ W, int ldb,
    const float* __restrict__ bias,
    float* __restrict__ C, int ldc,
    int N, int K) {
  const int lane = threadIdx.x & 31;
  const int wid  = threadIdx.x >> 5;
  const int m0 = blockIdx.x * 32 + (wid >> 2) * 16;
  const int n0 = blockIdx.y * 64 + (wid & 3) * 16;

  const bf16* arow = A + (size_t)(m0 + (lane & 15)) * lda;
  const bf16* brow = W + (size_t)(n0 + (lane & 15)) * ldb;

  v8f acc = {};
  for (int k0 = 0; k0 < K; k0 += 32)
    acc = wmma_bf(frag16(arow, k0, lane), frag16(brow, k0, lane), acc);

  const int n = n0 + (lane & 15);
  if (n < N) {
    const float bb = bias[n];
    const int mbase = m0 + ((lane >> 4) << 3);
#pragma unroll
    for (int v = 0; v < 8; ++v)
      C[(size_t)(mbase + v) * ldc + n] = acc[v] + bb;
  }
}

// ---------------------------------------------------------------------------
// Fused LSTM timestep: gates = xg[t] + h_prev @ W_hh^T ; nonlinearity fused.
// Wave tile: 32(batch) x 16(unit) x 4(gates) => per K-chunk 2 A-frags +
// 4 B-frags -> 8 WMMA.  Block 8 waves = 64(batch) x 64(unit).
// grid = (BAT/64, HID/64, ndir); block = 256.  W_hh is L2-resident: direct
// global loads, no barriers (latency-critical sequential chain).
// ---------------------------------------------------------------------------
struct StepArgs {
  const float* xg;     // [BAT, 4H] gate pre-activations for this timestep
  const bf16*  hprev;  // [BAT, ldh] previous hidden state (bf16)
  int          ldh;
  const bf16*  whh;    // [4H, HID] recurrent weights (bf16, row-major)
  float*       c;      // [BAT, HID] cell state (in/out, f32)
  bf16*        hout;   // [BAT, ldo] new hidden state (bf16)
  int          ldo;
};

__global__ __launch_bounds__(256) void k_lstm_step(StepArgs a0, StepArgs a1) {
  const StepArgs A = blockIdx.z ? a1 : a0;
  const int lane = threadIdx.x & 31;
  const int wid  = threadIdx.x >> 5;
  const int m0 = blockIdx.x * 64 + (wid >> 2) * 32;  // batch tile (32 rows)
  const int u0 = blockIdx.y * 64 + (wid & 3) * 16;   // hidden-unit tile

  const bf16* hrow0 = A.hprev + (size_t)(m0 + (lane & 15)) * A.ldh;
  const bf16* hrow1 = hrow0 + (size_t)16 * A.ldh;
  const int   wr    = u0 + (lane & 15);
  const bf16* wg0   = A.whh + (size_t)wr * HID;      // gate g row: + g*HID*HID

  v8f acc[2][4] = {};
  for (int k0 = 0; k0 < HID; k0 += 32) {
    const v16bf a0 = frag16(hrow0, k0, lane);
    const v16bf a1 = frag16(hrow1, k0, lane);
#pragma unroll
    for (int g = 0; g < 4; ++g) {
      const v16bf b = frag16(wg0 + (size_t)g * HID * HID, k0, lane);
      acc[0][g] = wmma_bf(a0, b, acc[0][g]);
      acc[1][g] = wmma_bf(a1, b, acc[1][g]);
    }
  }

  const int u  = u0 + (lane & 15);
  const int lm = (lane >> 4) << 3;
#pragma unroll
  for (int i = 0; i < 2; ++i) {
    const int mb = m0 + i * 16 + lm;
#pragma unroll
    for (int v = 0; v < 8; ++v) {
      const int m = mb + v;
      const size_t gi = (size_t)m * G4 + u;
      const float ig = sigm(A.xg[gi + 0 * HID] + acc[i][0][v]);
      const float fg = sigm(A.xg[gi + 1 * HID] + acc[i][1][v]);
      const float gg = tanhf(A.xg[gi + 2 * HID] + acc[i][2][v]);
      const float og = sigm(A.xg[gi + 3 * HID] + acc[i][3][v]);
      const size_t ci = (size_t)m * HID + u;
      const float cn = fg * A.c[ci] + ig * gg;
      A.c[ci] = cn;
      A.hout[(size_t)m * A.ldo + u] = (bf16)(og * tanhf(cn));
    }
  }
}

// ---------------------------------------------------------------------------
// Elementwise utility kernels
// ---------------------------------------------------------------------------
__global__ __launch_bounds__(256) void k_cvt_bf16(const float* __restrict__ s,
                                                  bf16* __restrict__ d, int n) {
  for (int i = blockIdx.x * 256 + threadIdx.x; i < n; i += gridDim.x * 256)
    d[i] = (bf16)s[i];
}

__global__ __launch_bounds__(256) void k_cvt_bf16_pad(const float* __restrict__ s,
                                                      bf16* __restrict__ d,
                                                      int rows, int cols, int padrows) {
  const int n = padrows * cols;
  for (int i = blockIdx.x * 256 + threadIdx.x; i < n; i += gridDim.x * 256) {
    const int r = i / cols;
    d[i] = (r < rows) ? (bf16)s[i] : (bf16)0.0f;
  }
}

__global__ __launch_bounds__(256) void k_zero_f32(float* p, int n) {
  for (int i = blockIdx.x * 256 + threadIdx.x; i < n; i += gridDim.x * 256) p[i] = 0.0f;
}

__global__ __launch_bounds__(256) void k_zero_bf16(bf16* p, int n) {
  for (int i = blockIdx.x * 256 + threadIdx.x; i < n; i += gridDim.x * 256) p[i] = (bf16)0.0f;
}

__global__ __launch_bounds__(256) void k_embed(const int* __restrict__ ids,
                                               const float* __restrict__ emb,
                                               bf16* __restrict__ out) {
  const int n = SB * EMB;
  for (int i = blockIdx.x * 256 + threadIdx.x; i < n; i += gridDim.x * 256) {
    const int sb = i / EMB, e = i - sb * EMB;
    out[i] = (bf16)emb[(size_t)ids[sb] * EMB + e];
  }
}

// ---------------------------------------------------------------------------
// Host orchestration
// ---------------------------------------------------------------------------
static inline int grd(int n) { int b = (n + 255) / 256; return b > 8192 ? 8192 : b; }

extern "C" void kernel_launch(void* const* d_in, const int* in_sizes, int n_in,
                              void* d_out, int out_size, void* d_ws, size_t ws_size,
                              hipStream_t stream) {
  (void)in_sizes; (void)n_in; (void)out_size; (void)ws_size;

  const int*   input_seq  = (const int*)d_in[0];
  const float* embedding  = (const float*)d_in[2];
  const float* enc_w_ih_f = (const float*)d_in[3];
  const float* enc_w_hh_f = (const float*)d_in[4];
  const float* enc_b_f    = (const float*)d_in[5];
  const float* enc_w_ih_b = (const float*)d_in[6];
  const float* enc_w_hh_b = (const float*)d_in[7];
  const float* enc_b_b    = (const float*)d_in[8];
  const float* dec_w_ih   = (const float*)d_in[9];
  const float* dec_w_hh   = (const float*)d_in[10];
  const float* dec_b      = (const float*)d_in[11];
  const float* out_w      = (const float*)d_in[12];
  const float* out_b      = (const float*)d_in[13];
  float*       preds      = (float*)d_out;

  // ---- workspace carve -----------------------------------------------------
  uint8_t* base = (uint8_t*)d_ws;
  size_t off = 0;
  auto take = [&](size_t bytes) -> void* {
    void* r = base + off;
    off += (bytes + 255) & ~(size_t)255;
    return r;
  };

  bf16* emb_bf      = (bf16*)take((size_t)SB * EMB * 2);
  bf16* w_ih_f_bf   = (bf16*)take((size_t)G4 * EMB * 2);
  bf16* w_hh_f_bf   = (bf16*)take((size_t)G4 * HID * 2);
  bf16* w_ih_b_bf   = (bf16*)take((size_t)G4 * EMB * 2);
  bf16* w_hh_b_bf   = (bf16*)take((size_t)G4 * HID * 2);
  bf16* dec_wih_bf  = (bf16*)take((size_t)G4 * 2 * HID * 2);
  bf16* dec_whh_bf  = (bf16*)take((size_t)G4 * HID * 2);
  bf16* out_w_bf    = (bf16*)take((size_t)VOCP * HID * 2);
  // encoded buffer: slots 0..SEQ+1; slot t+1 holds encoded[t]; slots 0 and
  // SEQ+1 are zero (h0 for fwd/bwd chains + decoder input 0).
  bf16* ENC         = (bf16*)take((size_t)(SEQ + 2) * BAT * 2 * HID * 2);
  bf16* dec_out_bf  = (bf16*)take((size_t)SB * HID * 2);
  bf16* zero_h      = (bf16*)take((size_t)BAT * HID * 2);
  float* xg_f       = (float*)take((size_t)SB * G4 * 4);
  float* xg_b       = (float*)take((size_t)SB * G4 * 4);
  float* c_f        = (float*)take((size_t)BAT * HID * 4);
  float* c_b        = (float*)take((size_t)BAT * HID * 4);
  float* c_d        = (float*)take((size_t)BAT * HID * 4);
  float* xg_d       = xg_f;  // encoder-fwd xg is dead before decoder runs

  // ---- precision conversion + gather --------------------------------------
  k_cvt_bf16<<<grd(G4 * EMB),     256, 0, stream>>>(enc_w_ih_f, w_ih_f_bf, G4 * EMB);
  k_cvt_bf16<<<grd(G4 * HID),     256, 0, stream>>>(enc_w_hh_f, w_hh_f_bf, G4 * HID);
  k_cvt_bf16<<<grd(G4 * EMB),     256, 0, stream>>>(enc_w_ih_b, w_ih_b_bf, G4 * EMB);
  k_cvt_bf16<<<grd(G4 * HID),     256, 0, stream>>>(enc_w_hh_b, w_hh_b_bf, G4 * HID);
  k_cvt_bf16<<<grd(G4 * 2 * HID), 256, 0, stream>>>(dec_w_ih,   dec_wih_bf, G4 * 2 * HID);
  k_cvt_bf16<<<grd(G4 * HID),     256, 0, stream>>>(dec_w_hh,   dec_whh_bf, G4 * HID);
  k_cvt_bf16_pad<<<grd(VOCP * HID), 256, 0, stream>>>(out_w, out_w_bf, VOC, HID, VOCP);
  k_embed<<<grd(SB * EMB), 256, 0, stream>>>(input_seq, embedding, emb_bf);

  // ---- zero-init state (workspace is poisoned by the harness) --------------
  k_zero_f32<<<grd(3 * BAT * HID), 256, 0, stream>>>(c_f, 3 * BAT * HID);
  k_zero_bf16<<<grd(BAT * 2 * HID), 256, 0, stream>>>(ENC, BAT * 2 * HID);
  k_zero_bf16<<<grd(BAT * 2 * HID), 256, 0, stream>>>(
      ENC + (size_t)(SEQ + 1) * BAT * 2 * HID, BAT * 2 * HID);
  k_zero_bf16<<<grd(BAT * HID), 256, 0, stream>>>(zero_h, BAT * HID);

  // ---- encoder input projections: xg = emb @ W_ih^T + b  [SB, 4H] ----------
  k_gemm_bias_big<<<dim3(SB / BM, G4 / BN), 256, 0, stream>>>(
      emb_bf, EMB, w_ih_f_bf, EMB, enc_b_f, xg_f, G4, EMB);
  k_gemm_bias_big<<<dim3(SB / BM, G4 / BN), 256, 0, stream>>>(
      emb_bf, EMB, w_ih_b_bf, EMB, enc_b_b, xg_b, G4, EMB);

  // ---- encoder recurrence: fwd (t asc) and bwd (t desc) in one launch ------
  for (int t = 0; t < SEQ; ++t) {
    StepArgs fa, ba;
    fa.xg    = xg_f + (size_t)t * BAT * G4;
    fa.hprev = ENC + (size_t)t * BAT * 2 * HID;            fa.ldh = 2 * HID;
    fa.whh   = w_hh_f_bf;
    fa.c     = c_f;
    fa.hout  = ENC + (size_t)(t + 1) * BAT * 2 * HID;      fa.ldo = 2 * HID;

    const int tb = SEQ - 1 - t;
    ba.xg    = xg_b + (size_t)tb * BAT * G4;
    ba.hprev = ENC + (size_t)(tb + 2) * BAT * 2 * HID + HID;  ba.ldh = 2 * HID;
    ba.whh   = w_hh_b_bf;
    ba.c     = c_b;
    ba.hout  = ENC + (size_t)(tb + 1) * BAT * 2 * HID + HID;  ba.ldo = 2 * HID;

    k_lstm_step<<<dim3(BAT / 64, HID / 64, 2), 256, 0, stream>>>(fa, ba);
  }

  // ---- decoder input projection: A = encoded shifted = ENC slots 0..SEQ-1 --
  k_gemm_bias_big<<<dim3(SB / BM, G4 / BN), 256, 0, stream>>>(
      ENC, 2 * HID, dec_wih_bf, 2 * HID, dec_b, xg_d, G4, 2 * HID);

  // ---- decoder recurrence --------------------------------------------------
  for (int t = 0; t < SEQ; ++t) {
    StepArgs da;
    da.xg    = xg_d + (size_t)t * BAT * G4;
    da.hprev = (t == 0) ? zero_h : dec_out_bf + (size_t)(t - 1) * BAT * HID;
    da.ldh   = HID;
    da.whh   = dec_whh_bf;
    da.c     = c_d;
    da.hout  = dec_out_bf + (size_t)t * BAT * HID;
    da.ldo   = HID;
    k_lstm_step<<<dim3(BAT / 64, HID / 64, 1), 256, 0, stream>>>(da, da);
  }

  // ---- vocab projection: preds[SB, 61] = dec_out @ out_w^T + out_b ---------
  k_gemm_bias_small<<<dim3(SB / 32, 1), 256, 0, stream>>>(
      dec_out_bf, HID, out_w_bf, HID, out_b, preds, VOC, VOC, HID);
}
